// MeshDiT_78623671320691
// MI455X (gfx1250) — 16-bit and 8-bit matrix operand order repaired and checked by reading, compiled, not run on hardware
//
#include <hip/hip_runtime.h>
#include <hip/hip_bf16.h>
#include <stdint.h>


typedef __attribute__((ext_vector_type(16))) _Float16 v16h;
typedef __attribute__((ext_vector_type(8)))  float    v8f;
typedef __attribute__((ext_vector_type(8)))  int      v8i;

#define D_      1024
#define H_      16
#define HD_     64
#define R_      16
#define I_      2816
#define B_      4
#define SIXD_   6144

static __device__ __forceinline__ float sigmoidf_(float x) { return 1.0f / (1.0f + __expf(-x)); }
static __device__ __forceinline__ float siluf_(float x)    { return x * sigmoidf_(x); }

union U16x8 { int4 v; _Float16 h[8]; };


__global__ void k_abs_partial(const float* __restrict__ w, long n, float* __restrict__ partial) {
    __shared__ float red[256];
    float s = 0.0f;
    for (long i = (long)blockIdx.x * blockDim.x + threadIdx.x; i < n; i += (long)gridDim.x * blockDim.x)
        s += fabsf(w[i]);
    red[threadIdx.x] = s; __syncthreads();
    for (int o = 128; o > 0; o >>= 1) { if ((int)threadIdx.x < o) red[threadIdx.x] += red[threadIdx.x + o]; __syncthreads(); }
    if (threadIdx.x == 0) partial[blockIdx.x] = red[0];
}

__global__ void k_abs_final(const float* __restrict__ partial, int np, long n, float* __restrict__ sc) {
    __shared__ float red[256];
    float s = 0.0f;
    for (int i = threadIdx.x; i < np; i += 256) s += partial[i];
    red[threadIdx.x] = s; __syncthreads();
    for (int o = 128; o > 0; o >>= 1) { if ((int)threadIdx.x < o) red[threadIdx.x] += red[threadIdx.x + o]; __syncthreads(); }
    if (threadIdx.x == 0) {
        float mean = red[0] / (float)n;
        float cm = fmaxf(mean, 1e-5f);
        sc[0] = 1.0f / cm;
        sc[1] = cm;
    }
}

__global__ void k_wquant(const float* __restrict__ w, const float* __restrict__ sc,
                         int8_t* __restrict__ q, long n) {
    float sw = sc[0];
    for (long i = (long)blockIdx.x * blockDim.x + threadIdx.x; i < n; i += (long)gridDim.x * blockDim.x) {
        float v = rintf(w[i] * sw);
        v = fminf(fmaxf(v, -1.0f), 1.0f);
        q[i] = (int8_t)v;
    }
}


__global__ void k_rms_quant(const float* __restrict__ x, int8_t* __restrict__ xq,
                            float* __restrict__ arcp, int K) {
    long row = blockIdx.x;
    const float* xr = x + row * (long)K;
    __shared__ float red[256];
    __shared__ float srn, ssq;
    float ss = 0.0f;
    for (int k = threadIdx.x; k < K; k += 256) { float v = xr[k]; ss += v * v; }
    red[threadIdx.x] = ss; __syncthreads();
    for (int o = 128; o > 0; o >>= 1) { if ((int)threadIdx.x < o) red[threadIdx.x] += red[threadIdx.x + o]; __syncthreads(); }
    if (threadIdx.x == 0) srn = rsqrtf(red[0] / (float)K + 1e-6f);
    __syncthreads();
    float rn = srn;
    float mx = 0.0f;
    for (int k = threadIdx.x; k < K; k += 256) mx = fmaxf(mx, fabsf(xr[k] * rn));
    red[threadIdx.x] = mx; __syncthreads();
    for (int o = 128; o > 0; o >>= 1) { if ((int)threadIdx.x < o) red[threadIdx.x] = fmaxf(red[threadIdx.x], red[threadIdx.x + o]); __syncthreads(); }
    if (threadIdx.x == 0) {
        float s = 127.0f / fmaxf(red[0], 1e-5f);
        ssq = s;
        arcp[row] = 1.0f / s;
    }
    __syncthreads();
    float s = ssq;
    for (int k = threadIdx.x; k < K; k += 256) {
        float q = rintf(xr[k] * rn * s);
        q = fminf(fmaxf(q, -128.0f), 127.0f);
        xq[row * (long)K + k] = (int8_t)q;
    }
}


__global__ void k_gemm_i8(const int8_t* __restrict__ xq, const int8_t* __restrict__ wq,
                          const float* __restrict__ arcp, const float* __restrict__ wsc,
                          const float* __restrict__ bias, float* __restrict__ out,
                          int M, int N, int K) {
    int lane = threadIdx.x & 31;
    int wv   = threadIdx.x >> 5;
    int half = lane >> 4;
    int l15  = lane & 15;
    int m0 = blockIdx.x * 16;
    int n0 = (blockIdx.y * 4 + wv) * 32;
    int mA = m0 + l15;
    if (mA >= M) mA = M - 1;
    const int8_t* arow  = xq + (long)mA * K + 16 * half;
    const int8_t* brow0 = wq + (long)(n0 + l15) * K + 16 * half;
    const int8_t* brow1 = wq + (long)(n0 + 16 + l15) * K + 16 * half;

    v8i acc0 = {}, acc1 = {};
    for (int k0 = 0; k0 < K; k0 += 64) {
        __builtin_prefetch(arow  + k0 + 64, 0, 3);
        __builtin_prefetch(brow0 + k0 + 64, 0, 3);
        __builtin_prefetch(brow1 + k0 + 64, 0, 3);
        v8i a, b0, b1;
#pragma unroll
        for (int jj = 0; jj < 4; jj++) {
            int2 t = *(const int2*)(arow + k0 + 8 * (jj & 1) + 32 * (jj >> 1));
            a[2 * jj] = t.x; a[2 * jj + 1] = t.y;
        }
        {
            int4 t0 = *(const int4*)(brow0 + k0);
            int4 t1 = *(const int4*)(brow0 + k0 + 32);
            b0[0] = t0.x; b0[1] = t0.y; b0[2] = t0.z; b0[3] = t0.w;
            b0[4] = t1.x; b0[5] = t1.y; b0[6] = t1.z; b0[7] = t1.w;
            int4 u0 = *(const int4*)(brow1 + k0);
            int4 u1 = *(const int4*)(brow1 + k0 + 32);
            b1[0] = u0.x; b1[1] = u0.y; b1[2] = u0.z; b1[3] = u0.w;
            b1[4] = u1.x; b1[5] = u1.y; b1[6] = u1.z; b1[7] = u1.w;
        }
        acc0 = __builtin_amdgcn_wmma_i32_16x16x64_iu8(true, a, true, b0, acc0, false, false);
        acc1 = __builtin_amdgcn_wmma_i32_16x16x64_iu8(true, a, true, b1, acc1, false, false);
    }
    float wr = wsc[1];
#pragma unroll
    for (int r = 0; r < 8; r++) {
        int m = m0 + r + 8 * half;
        if (m < M) {
            float sc = arcp[m] * wr;
            int n = n0 + l15;
            float v0 = (float)acc0[r] * sc;
            float v1 = (float)acc1[r] * sc;
            if (bias) { v0 += bias[n]; v1 += bias[n + 16]; }
            out[(long)m * N + n]      = v0;
            out[(long)m * N + n + 16] = v1;
        }
    }
}

__global__ void k_fpgemm_nt(const float* __restrict__ A, const float* __restrict__ Bt,
                            float* __restrict__ out, int M, int N, int K) {
    long idx = (long)blockIdx.x * blockDim.x + threadIdx.x;
    if (idx >= (long)M * N) return;
    int m = (int)(idx / N), n = (int)(idx % N);
    const float* a = A + (long)m * K;
    const float* b = Bt + (long)n * K;
    float s = 0.0f;
    for (int k = 0; k < K; k++) s += a[k] * b[k];
    out[idx] = s;
}


__global__ void k_silu(const float* __restrict__ in, float* __restrict__ out, long n) {
    long i = (long)blockIdx.x * blockDim.x + threadIdx.x;
    if (i < n) out[i] = siluf_(in[i]);
}

__global__ void k_layernorm(const float* __restrict__ x, const float* __restrict__ g,
                            const float* __restrict__ b, float* __restrict__ out, int K) {
    long row = blockIdx.x;
    const float* xr = x + row * (long)K;
    __shared__ float red[256];
    __shared__ float smean, srv;
    float s = 0.0f;
    for (int k = threadIdx.x; k < K; k += 256) s += xr[k];
    red[threadIdx.x] = s; __syncthreads();
    for (int o = 128; o > 0; o >>= 1) { if ((int)threadIdx.x < o) red[threadIdx.x] += red[threadIdx.x + o]; __syncthreads(); }
    if (threadIdx.x == 0) smean = red[0] / (float)K;
    __syncthreads();
    float mean = smean;
    float v = 0.0f;
    for (int k = threadIdx.x; k < K; k += 256) { float d = xr[k] - mean; v += d * d; }
    red[threadIdx.x] = v; __syncthreads();
    for (int o = 128; o > 0; o >>= 1) { if ((int)threadIdx.x < o) red[threadIdx.x] += red[threadIdx.x + o]; __syncthreads(); }
    if (threadIdx.x == 0) srv = rsqrtf(red[0] / (float)K + 1e-5f);
    __syncthreads();
    float rv = srv;
    for (int k = threadIdx.x; k < K; k += 256)
        out[row * (long)K + k] = (xr[k] - mean) * rv * g[k] + b[k];
}

__global__ void k_modulate(const float* __restrict__ x, const float* __restrict__ mod,
                           int shift_off, int scale_off, float* __restrict__ out,
                           int Nseq, long total) {
    long idx = (long)blockIdx.x * blockDim.x + threadIdx.x;
    if (idx >= total) return;
    int k = (int)(idx % D_);
    long m = idx / D_;
    int bidx = (int)(m / Nseq);
    const float* mrow = mod + (long)bidx * SIXD_;
    out[idx] = x[idx] * (1.0f + mrow[scale_off + k]) + mrow[shift_off + k];
}

__global__ void k_resid(const float* __restrict__ base, const float* __restrict__ val,
                        const float* __restrict__ mod, int gate_off, float* __restrict__ out,
                        int Nseq, long total) {
    long idx = (long)blockIdx.x * blockDim.x + threadIdx.x;
    if (idx >= total) return;
    int k = (int)(idx % D_);
    long m = idx / D_;
    int bidx = (int)(m / Nseq);
    out[idx] = base[idx] + mod[(long)bidx * SIXD_ + gate_off + k] * val[idx];
}


__global__ void k_attn_prep(const float* __restrict__ qkv, const float* __restrict__ delta,
                            const float* __restrict__ mag,
                            _Float16* __restrict__ Q, _Float16* __restrict__ K,
                            _Float16* __restrict__ Vt, int Nseq, long total) {
    long idx = (long)blockIdx.x * blockDim.x + threadIdx.x;
    if (idx >= total) return;
    int d = (int)(idx & 63);
    long t = idx >> 6;
    int n = (int)(t % Nseq); t /= Nseq;
    int h = (int)(t % H_);
    int b = (int)(t / H_);
    long row = (long)b * Nseq + n;
    float qv = qkv[row * (3 * D_) + h * HD_ + d] + mag[h] * delta[row * D_ + h * HD_ + d];
    float kv = qkv[row * (3 * D_) + D_ + h * HD_ + d];
    float vv = qkv[row * (3 * D_) + 2 * D_ + h * HD_ + d];
    long bh = (long)b * H_ + h;
    Q[(bh * Nseq + n) * HD_ + d]  = (_Float16)qv;
    K[(bh * Nseq + n) * HD_ + d]  = (_Float16)kv;
    Vt[(bh * HD_ + d) * Nseq + n] = (_Float16)vv;
}

__global__ void __launch_bounds__(32)
k_attn(const _Float16* __restrict__ Qg, const _Float16* __restrict__ Kg,
       const _Float16* __restrict__ Vg, float* __restrict__ out, int Nseq) {
    int lane = threadIdx.x;
    int half = lane >> 4;
    int l15  = lane & 15;
    long bh  = blockIdx.y;
    int q0   = blockIdx.x * 16;
    const _Float16* Q  = Qg + bh * (long)Nseq * HD_;
    const _Float16* Km = Kg + bh * (long)Nseq * HD_;
    const _Float16* Vm = Vg + bh * (long)HD_ * Nseq;

    __shared__ float    sS[16 * 32];
    __shared__ _Float16 sP[16 * 32];
    __shared__ float    smx[16], sml[16], ssc[16];
    if (lane < 16) { smx[lane] = -3.0e38f; sml[lane] = 0.0f; ssc[lane] = 1.0f; }

    v16h qa[2];
    int qrow = q0 + l15;
#pragma unroll
    for (int s = 0; s < 2; s++) {
        U16x8 u0, u1;
        u0.v = *(const int4*)(Q + (long)qrow * HD_ + s * 32 + 8 * half);
        u1.v = *(const int4*)(Q + (long)qrow * HD_ + s * 32 + 16 + 8 * half);
#pragma unroll
        for (int t = 0; t < 8; t++) { qa[s][t] = u0.h[t]; qa[s][8 + t] = u1.h[t]; }
    }
    v8f o0 = {}, o1 = {}, o2 = {}, o3 = {};
    __syncthreads();

    for (int kc = 0; kc < Nseq; kc += 32) {
#pragma unroll
        for (int t = 0; t < 2; t++) {
            v8f sa = {};
#pragma unroll
            for (int s = 0; s < 2; s++) {
                v16h kb;
                int key = kc + t * 16 + l15;
                U16x8 u0, u1;
                u0.v = *(const int4*)(Km + (long)key * HD_ + s * 32 + 8 * half);
                u1.v = *(const int4*)(Km + (long)key * HD_ + s * 32 + 16 + 8 * half);
#pragma unroll
                for (int e = 0; e < 8; e++) { kb[e] = u0.h[e]; kb[8 + e] = u1.h[e]; }
                sa = __builtin_amdgcn_wmma_f32_16x16x32_f16(false, qa[s], false, kb,
                                                            (short)0, sa, false, false);
            }
#pragma unroll
            for (int r = 0; r < 8; r++) {
                int m = r + 8 * half;
                sS[m * 32 + t * 16 + l15] = sa[r] * 0.125f;
            }
        }
        __syncthreads();
        if (lane < 16) {
            int m = lane;
            float mold = smx[m], cm = mold;
            for (int cix = 0; cix < 32; cix++) cm = fmaxf(cm, sS[m * 32 + cix]);
            float sc = __expf(mold - cm);
            float ls = sml[m] * sc;
            for (int cix = 0; cix < 32; cix++) {
                float p = __expf(sS[m * 32 + cix] - cm);
                sP[m * 32 + cix] = (_Float16)p;
                ls += p;
            }
            smx[m] = cm; sml[m] = ls; ssc[m] = sc;
        }
        __syncthreads();
#pragma unroll
        for (int r = 0; r < 8; r++) {
            float sc = ssc[r + 8 * half];
            o0[r] *= sc; o1[r] *= sc; o2[r] *= sc; o3[r] *= sc;
        }
        v16h pa;
        {
            U16x8 u0, u1;
            u0.v = *(const int4*)(&sP[l15 * 32 + 8 * half]);
            u1.v = *(const int4*)(&sP[l15 * 32 + 16 + 8 * half]);
#pragma unroll
            for (int e = 0; e < 8; e++) { pa[e] = u0.h[e]; pa[8 + e] = u1.h[e]; }
        }
#pragma unroll
        for (int g = 0; g < 4; g++) {
            v16h vb;
            int hd = g * 16 + l15;
            U16x8 u0, u1;
            u0.v = *(const int4*)(Vm + (long)hd * Nseq + kc + 8 * half);
            u1.v = *(const int4*)(Vm + (long)hd * Nseq + kc + 16 + 8 * half);
#pragma unroll
            for (int e = 0; e < 8; e++) { vb[e] = u0.h[e]; vb[8 + e] = u1.h[e]; }
            v8f* op = (g == 0) ? &o0 : (g == 1) ? &o1 : (g == 2) ? &o2 : &o3;
            *op = __builtin_amdgcn_wmma_f32_16x16x32_f16(false, pa, false, vb,
                                                         (short)0, *op, false, false);
        }
        __syncthreads();
    }
    int b = (int)(bh / H_);
    int h = (int)(bh % H_);
#pragma unroll
    for (int r = 0; r < 8; r++) {
        int m = r + 8 * half;
        float inv = 1.0f / sml[m];
        long row = (long)b * Nseq + q0 + m;
        float* orow = out + row * D_ + h * HD_;
        orow[0 * 16 + l15] = o0[r] * inv;
        orow[1 * 16 + l15] = o1[r] * inv;
        orow[2 * 16 + l15] = o2[r] * inv;
        orow[3 * 16 + l15] = o3[r] * inv;
    }
}


__global__ void k_hgrn_pre(float* __restrict__ F, float* __restrict__ I, long n) {
    long i = (long)blockIdx.x * blockDim.x + threadIdx.x;
    if (i >= n) return;
    float f = sigmoidf_(F[i]);
    float iv = siluf_(I[i]) * (1.0f - f);
    F[i] = f; I[i] = iv;
}

__global__ void k_scan(const float* __restrict__ iv, const float* __restrict__ fv,
                       float* __restrict__ hs, int Nseq) {
    int ch = blockIdx.x * blockDim.x + threadIdx.x;
    if (ch >= B_ * D_) return;
    int b = ch / D_, c = ch % D_;
    float h = 0.0f;
    for (int n = 0; n < Nseq; n++) {
        long idx = ((long)b * Nseq + n) * D_ + c;
        h = fv[idx] * h + iv[idx];
        hs[idx] = h;
    }
}

__global__ void k_hgrn_post(const float* __restrict__ hs, const float* __restrict__ G,
                            const float* __restrict__ gn, float* __restrict__ o, int K) {
    long row = blockIdx.x;
    const float* xr = hs + row * (long)K;
    __shared__ float red[256];
    __shared__ float srn;
    float ss = 0.0f;
    for (int k = threadIdx.x; k < K; k += 256) { float v = xr[k]; ss += v * v; }
    red[threadIdx.x] = ss; __syncthreads();
    for (int of = 128; of > 0; of >>= 1) { if ((int)threadIdx.x < of) red[threadIdx.x] += red[threadIdx.x + of]; __syncthreads(); }
    if (threadIdx.x == 0) srn = rsqrtf(red[0] / (float)K + 1e-6f);
    __syncthreads();
    float rn = srn;
    for (int k = threadIdx.x; k < K; k += 256) {
        float g = G[row * (long)K + k];
        o[row * (long)K + k] = xr[k] * rn * gn[k] * g * sigmoidf_(g);
    }
}


__global__ void k_mlp_gate(const float* __restrict__ Y, float* __restrict__ z, long total) {
    long idx = (long)blockIdx.x * blockDim.x + threadIdx.x;
    if (idx >= total) return;
    long m = idx / I_;
    int k = (int)(idx % I_);
    float g = Y[m * (2L * I_) + k];
    float v = Y[m * (2L * I_) + I_ + k];
    z[idx] = siluf_(g) * v;
}


struct QW { const int8_t* q; const float* s; };

static inline long cdiv(long a, long b) { return (a + b - 1) / b; }

static void run_stream(const float* xin, float* xout, int Nseq,
                       const float* ln_g, const float* ln_b,
                       QW q_qkv, const float* mag, const float* la, const float* lb, QW q_wp,
                       QW q_wi, QW q_wf, QW q_wg, QW q_wo, const float* gn,
                       QW q_mg, QW q_md,
                       const float* mod,
                       float* ln, int8_t* xq, float* arcp, char* arena,
                       hipStream_t stream) {
    const int M = B_ * Nseq;
    const long MD = (long)M * D_;
    size_t off;
    auto A = [&](size_t b) -> char* { char* p = arena + off; off = (off + b + 255) & ~(size_t)255; return p; };

    k_layernorm<<<M, 256, 0, stream>>>(xin, ln_g, ln_b, ln, D_);

    off = 0;
    float*    qkv   = (float*)A(MD * 3 * 4);
    float*    delta = (float*)A(MD * 4);
    float*    t1    = (float*)A((size_t)M * R_ * 4);
    _Float16* Qf    = (_Float16*)A(MD * 2);
    _Float16* Kf    = (_Float16*)A(MD * 2);
    _Float16* Vt    = (_Float16*)A(MD * 2);
    float*    attn  = (float*)A(MD * 4);
    float*    proj  = (float*)A(MD * 4);

    k_rms_quant<<<M, 256, 0, stream>>>(ln, xq, arcp, D_);
    k_gemm_i8<<<dim3((M + 15) / 16, (3 * D_) / 128), 128, 0, stream>>>(
        xq, q_qkv.q, arcp, q_qkv.s, nullptr, qkv, M, 3 * D_, D_);
    k_fpgemm_nt<<<(unsigned)cdiv((long)M * R_, 256), 256, 0, stream>>>(ln, la, t1, M, R_, D_);
    k_fpgemm_nt<<<(unsigned)cdiv(MD, 256), 256, 0, stream>>>(t1, lb, delta, M, D_, R_);
    long npre = (long)B_ * H_ * Nseq * HD_;
    k_attn_prep<<<(unsigned)cdiv(npre, 256), 256, 0, stream>>>(qkv, delta, mag, Qf, Kf, Vt, Nseq, npre);
    k_attn<<<dim3(Nseq / 16, B_ * H_), 32, 0, stream>>>(Qf, Kf, Vt, attn, Nseq);
    k_rms_quant<<<M, 256, 0, stream>>>(attn, xq, arcp, D_);
    k_gemm_i8<<<dim3((M + 15) / 16, D_ / 128), 128, 0, stream>>>(
        xq, q_wp.q, arcp, q_wp.s, nullptr, proj, M, D_, D_);
    k_resid<<<(unsigned)cdiv(MD, 256), 256, 0, stream>>>(xin, proj, mod, 2 * D_, xout, Nseq, MD);

    off = 0;
    float* xm   = (float*)A(MD * 4);
    float* Fb   = (float*)A(MD * 4);
    float* Ib   = (float*)A(MD * 4);
    float* Gb   = (float*)A(MD * 4);
    float* hsb  = (float*)A(MD * 4);
    float* ob   = (float*)A(MD * 4);
    float* hout = (float*)A(MD * 4);

    k_modulate<<<(unsigned)cdiv(MD, 256), 256, 0, stream>>>(ln, mod, 0, D_, xm, Nseq, MD);
    k_rms_quant<<<M, 256, 0, stream>>>(xm, xq, arcp, D_);
    k_gemm_i8<<<dim3((M + 15) / 16, D_ / 128), 128, 0, stream>>>(xq, q_wi.q, arcp, q_wi.s, nullptr, Ib, M, D_, D_);
    k_gemm_i8<<<dim3((M + 15) / 16, D_ / 128), 128, 0, stream>>>(xq, q_wf.q, arcp, q_wf.s, nullptr, Fb, M, D_, D_);
    k_gemm_i8<<<dim3((M + 15) / 16, D_ / 128), 128, 0, stream>>>(xq, q_wg.q, arcp, q_wg.s, nullptr, Gb, M, D_, D_);
    k_hgrn_pre<<<(unsigned)cdiv(MD, 256), 256, 0, stream>>>(Fb, Ib, MD);
    k_scan<<<(B_ * D_ + 255) / 256, 256, 0, stream>>>(Ib, Fb, hsb, Nseq);
    k_hgrn_post<<<M, 256, 0, stream>>>(hsb, Gb, gn, ob, D_);
    k_rms_quant<<<M, 256, 0, stream>>>(ob, xq, arcp, D_);
    k_gemm_i8<<<dim3((M + 15) / 16, D_ / 128), 128, 0, stream>>>(xq, q_wo.q, arcp, q_wo.s, nullptr, hout, M, D_, D_);
    k_resid<<<(unsigned)cdiv(MD, 256), 256, 0, stream>>>(xout, hout, mod, 2 * D_, xout, Nseq, MD);

    off = 0;
    float* xm2 = (float*)A(MD * 4);
    float* Yb  = (float*)A((long)M * 2 * I_ * 4);
    float* zb  = (float*)A((long)M * I_ * 4);

    k_layernorm<<<M, 256, 0, stream>>>(xout, ln_g, ln_b, xm2, D_);
    k_modulate<<<(unsigned)cdiv(MD, 256), 256, 0, stream>>>(xm2, mod, 3 * D_, 4 * D_, xm2, Nseq, MD);
    k_rms_quant<<<M, 256, 0, stream>>>(xm2, xq, arcp, D_);
    k_gemm_i8<<<dim3((M + 15) / 16, (2 * I_) / 128), 128, 0, stream>>>(
        xq, q_mg.q, arcp, q_mg.s, nullptr, Yb, M, 2 * I_, D_);
    long zt = (long)M * I_;
    k_mlp_gate<<<(unsigned)cdiv(zt, 256), 256, 0, stream>>>(Yb, zb, zt);
    k_rms_quant<<<M, 256, 0, stream>>>(zb, xq, arcp, I_);
    k_gemm_i8<<<dim3((M + 15) / 16, D_ / 128), 128, 0, stream>>>(
        xq, q_md.q, arcp, q_md.s, nullptr, xm2, M, D_, I_);
    k_resid<<<(unsigned)cdiv(MD, 256), 256, 0, stream>>>(xout, xm2, mod, 5 * D_, xout, Nseq, MD);
}

extern "C" void kernel_launch(void* const* d_in, const int* in_sizes, int n_in,
                              void* d_out, int out_size, void* d_ws, size_t ws_size,
                              hipStream_t stream) {
    (void)in_sizes; (void)n_in; (void)out_size; (void)ws_size;
    const float* x       = (const float*)d_in[0];
    const float* y       = (const float*)d_in[1];
    const float* c       = (const float*)d_in[2];
    const float* ada_w   = (const float*)d_in[3];
    const float* ada_b   = (const float*)d_in[4];
    const float* lnx_g   = (const float*)d_in[5];
    const float* lnx_b   = (const float*)d_in[6];
    const float* lny_g   = (const float*)d_in[7];
    const float* lny_b   = (const float*)d_in[8];
    const float* cax_wqkv= (const float*)d_in[9];
    const float* cax_mag = (const float*)d_in[10];
    const float* cax_la  = (const float*)d_in[11];
    const float* cax_lb  = (const float*)d_in[12];
    const float* cax_wp  = (const float*)d_in[13];
    const float* cay_wqkv= (const float*)d_in[14];
    const float* cay_mag = (const float*)d_in[15];
    const float* cay_la  = (const float*)d_in[16];
    const float* cay_lb  = (const float*)d_in[17];
    const float* cay_wp  = (const float*)d_in[18];
    const float* ax_wi   = (const float*)d_in[19];
    const float* ax_wf   = (const float*)d_in[20];
    const float* ax_wg   = (const float*)d_in[21];
    const float* ax_wo   = (const float*)d_in[22];
    const float* ax_gn   = (const float*)d_in[23];
    const float* ay_wi   = (const float*)d_in[24];
    const float* ay_wf   = (const float*)d_in[25];
    const float* ay_wg   = (const float*)d_in[26];
    const float* ay_wo   = (const float*)d_in[27];
    const float* ay_gn   = (const float*)d_in[28];
    const float* mx_wg   = (const float*)d_in[29];
    const float* mx_wd   = (const float*)d_in[30];
    const float* my_wg   = (const float*)d_in[31];
    const float* my_wd   = (const float*)d_in[32];

    float* out_x = (float*)d_out;
    float* out_y = out_x + (size_t)B_ * 1024 * D_;

    char* base = (char*)d_ws;
    size_t off = 0;
    auto alloc = [&](size_t n) -> void* { void* p = base + off; off = (off + n + 255) & ~(size_t)255; return p; };

    float* partial = (float*)alloc(256 * sizeof(float));
    float* scales  = (float*)alloc(64 * sizeof(float));
    int sci = 0;
    auto quantw = [&](const float* w, long n) -> QW {
        int8_t* q = (int8_t*)alloc((size_t)n);
        float* sc = scales + 2 * (sci++);
        k_abs_partial<<<256, 256, 0, stream>>>(w, n, partial);
        k_abs_final<<<1, 256, 0, stream>>>(partial, 256, n, sc);
        long blocks = cdiv(n, 256); if (blocks > 8192) blocks = 8192;
        k_wquant<<<(unsigned)blocks, 256, 0, stream>>>(w, sc, q, n);
        QW r; r.q = q; r.s = sc; return r;
    };

    QW q_ada   = quantw(ada_w,    (long)SIXD_ * D_);
    QW qx_qkv  = quantw(cax_wqkv, 3L * D_ * D_);
    QW qx_wp   = quantw(cax_wp,   (long)D_ * D_);
    QW qy_qkv  = quantw(cay_wqkv, 3L * D_ * D_);
    QW qy_wp   = quantw(cay_wp,   (long)D_ * D_);
    QW qx_wi   = quantw(ax_wi,    (long)D_ * D_);
    QW qx_wf   = quantw(ax_wf,    (long)D_ * D_);
    QW qx_wg   = quantw(ax_wg,    (long)D_ * D_);
    QW qx_wo   = quantw(ax_wo,    (long)D_ * D_);
    QW qy_wi   = quantw(ay_wi,    (long)D_ * D_);
    QW qy_wf   = quantw(ay_wf,    (long)D_ * D_);
    QW qy_wg   = quantw(ay_wg,    (long)D_ * D_);
    QW qy_wo   = quantw(ay_wo,    (long)D_ * D_);
    QW qx_mg   = quantw(mx_wg,    2L * I_ * D_);
    QW qx_md   = quantw(mx_wd,    (long)D_ * I_);
    QW qy_mg   = quantw(my_wg,    2L * I_ * D_);
    QW qy_md   = quantw(my_wd,    (long)D_ * I_);

    float*  mod   = (float*)alloc((size_t)B_ * SIXD_ * 4);
    float*  ln    = (float*)alloc((size_t)B_ * 1024 * D_ * 4);
    int8_t* xq    = (int8_t*)alloc((size_t)B_ * 1024 * I_);
    float*  arcp  = (float*)alloc((size_t)B_ * 1024 * 4);
    float*  cbuf  = (float*)alloc((size_t)B_ * D_ * 4);
    char*   arena = (char*)alloc((size_t)160 * 1024 * 1024);

    k_silu<<<(B_ * D_ + 255) / 256, 256, 0, stream>>>(c, cbuf, (long)B_ * D_);
    k_rms_quant<<<B_, 256, 0, stream>>>(cbuf, xq, arcp, D_);
    k_gemm_i8<<<dim3(1, SIXD_ / 128), 128, 0, stream>>>(
        xq, q_ada.q, arcp, q_ada.s, ada_b, mod, B_, SIXD_, D_);

    run_stream(x, out_x, 1024, lnx_g, lnx_b,
               qx_qkv, cax_mag, cax_la, cax_lb, qx_wp,
               qx_wi, qx_wf, qx_wg, qx_wo, ax_gn,
               qx_mg, qx_md, mod, ln, xq, arcp, arena, stream);

    run_stream(y, out_y, 256, lny_g, lny_b,
               qy_qkv, cay_mag, cay_la, cay_lb, qy_wp,
               qy_wi, qy_wf, qy_wg, qy_wo, ay_gn,
               qy_mg, qy_md, mod, ln, xq, arcp, arena, stream);
}
